// SpatialGATAttention_60490319397015
// MI455X (gfx1250) — compile-verified, matrix operand order checked by reading, not run on hardware
//
#include <hip/hip_runtime.h>
#include <stdint.h>

// x:(8,64,32,400)  BT=256 graphs, N=400 nodes, C=64 ch, H=2 heads, D=32
// grid = 256 workgroups (one graph each), block = 256 threads = 8 waves (wave32)

typedef __attribute__((ext_vector_type(16))) __bf16 v16bf;
typedef __attribute__((ext_vector_type(8)))  float  v8f;

union BF16x16 { v16bf v; uint32_t u[8]; };
union F32x8   { v8f   v; float    f[8]; };

__device__ __forceinline__ uint32_t f2bf1(float f) {
  return (__float_as_uint(f) + 0x8000u) >> 16;          // round-to-nearest
}
// pack two f32 -> packed bf16 pair with one v_perm_b32 (3 VALU total)
__device__ __forceinline__ uint32_t pkbf(float lo, float hi) {
  return __builtin_amdgcn_perm(__float_as_uint(hi) + 0x8000u,
                               __float_as_uint(lo) + 0x8000u, 0x07060302u);
}
__device__ __forceinline__ float bf2f(uint32_t h) { return __uint_as_float(h << 16); }

// ---- LDS layout (bytes). Union region reused across phases. ----
constexpr int RSX  = 72;                        // xr row stride (bf16), 144B
constexpr int RSVT = 424;                       // vt row stride (bf16), 848B
constexpr int STAGE_OFF = 25600;                // after mask words (400*16*4)
constexpr int REGU      = 58880;                // max(xr 57600, mask + stage 8*16*65*4)
constexpr int VT_OFF    = REGU;                 // 64*848 = 54272
constexpr int WV_OFF    = VT_OFF + 64 * RSVT * 2;    // 113152 ; 64*144 = 9216
constexpr int SSRC_OFF  = WV_OFF + 64 * RSX * 2;     // 122368 ; [h][400] f32
constexpr int SDST_OFF  = SSRC_OFF + 800 * 4;        // 125568 ; [h][416] f32 (padded)
constexpr int ROWM_OFF  = SDST_OFF + 2 * 416 * 4;    // 128896 ; [h][400] f32
constexpr int WEFF_OFF  = ROWM_OFF + 800 * 4;        // 132096 ; 256 f32
constexpr int MAH_OFF   = WEFF_OFF + 256 * 4;        // 133120 ; 2 f32 (max sdst per head)
constexpr int SMEM_BYTES = MAH_OFF + 32;             // 133152

__global__ __launch_bounds__(256) void gat_fused(
    const float* __restrict__ x,  const float* __restrict__ Wq,
    const float* __restrict__ Wk, const float* __restrict__ Wv,
    const float* __restrict__ a_src, const float* __restrict__ a_dst,
    const float* __restrict__ gamma, const float* __restrict__ beta,
    const int*   __restrict__ gso,   float* __restrict__ out)
{
  extern __shared__ char smem[];
  uint16_t* xr    = (uint16_t*)(smem);               // phases 0-1
  uint32_t* maskw = (uint32_t*)(smem);               // phases 2+ (aliases dead xr)
  float*    stage = (float*)(smem + STAGE_OFF);      // phase 4 LN staging
  uint16_t* vt    = (uint16_t*)(smem + VT_OFF);      // V^T [ch][j] bf16
  uint16_t* wvb   = (uint16_t*)(smem + WV_OFF);      // Wv bf16 [e][c]
  float*    ssrc  = (float*)(smem + SSRC_OFF);       // [h*400+n]
  float*    sdstp = (float*)(smem + SDST_OFF);       // [h*416+j], j 400..415 zeroed
  float*    rowm  = (float*)(smem + ROWM_OFF);       // [h*400+i]
  float*    weff  = (float*)(smem + WEFF_OFF);       // [0:128)=q_eff, [128:256)=k_eff
  float*    maH   = (float*)(smem + MAH_OFF);        // max_j sdst[h][j]

  const int g    = blockIdx.x;          // bt index
  const int b    = g >> 5;              // T = 32
  const int t    = g & 31;
  const int tid  = threadIdx.x;
  const int wave = tid >> 5;
  const int lane = tid & 31;
  const int l15  = lane & 15;
  const int half = lane >> 4;

  // ---- Phase 0a: effective attention vectors  weff_q[h][c] = sum_d a_src[h,d]*Wq[h*32+d, c]
  {
    const int  h  = (tid >> 6) & 1;
    const int  c  = tid & 63;
    const float* av = (tid < 128) ? a_src : a_dst;
    const float* W  = (tid < 128) ? Wq    : Wk;
    float acc = 0.f;
    #pragma unroll
    for (int d = 0; d < 32; ++d) acc += av[h * 32 + d] * W[(h * 32 + d) * 64 + c];
    weff[(tid < 128 ? 0 : 128) + h * 64 + c] = acc;
  }
  // ---- Phase 0b: Wv -> bf16 LDS (B operand of V-GEMM: B[c][e] = Wv[e][c])
  for (int k = tid; k < 64 * 64; k += 256) {
    int e = k >> 6, c = k & 63;
    wvb[e * RSX + c] = (uint16_t)f2bf1(Wv[e * 64 + c]);
  }
  // ---- Phase 0c: stage x slice -> xr[n][c] bf16 (global reads coalesced over n)
  {
    const float* xg = x + ((size_t)b * 64) * 12800 + (size_t)t * 400; // (c,n) at xg[c*12800+n]
    for (int k = tid; k < 64 * 400; k += 256) {
      int c = k / 400, n = k - c * 400;
      xr[n * RSX + c] = (uint16_t)f2bf1(xg[c * 12800 + n]);
    }
    for (int p = tid; p < 400; p += 256) __builtin_prefetch(gso + p * 400, 0, 1);
  }
  // ---- Phase 0d: zero vt padding columns j=400..415 and sdst padding
  for (int k = tid; k < 64 * 16; k += 256) {
    int e = k >> 4;
    vt[e * RSVT + 400 + (k & 15)] = 0;
  }
  if (tid < 32) sdstp[(tid >> 4) * 416 + 400 + (tid & 15)] = 0.f;
  __syncthreads();

  // ---- Phase 1: V = xr @ Wv^T via v_wmma_f32_16x16x32_bf16; store V^T[ch][j] bf16
  for (int tile = wave; tile < 100; tile += 8) {      // 25 n-tiles x 4 e-tiles
    int nt = tile >> 2, et = tile & 3;
    int nb = nt * 16,   eb = et * 16;
    int arow = nb + l15, bcol = eb + l15;
    F32x8 acc;
    #pragma unroll
    for (int i = 0; i < 8; ++i) acc.f[i] = 0.f;
    #pragma unroll
    for (int kk = 0; kk < 2; ++kk) {
      BF16x16 A, Bm;
      uint4 a0 = *(const uint4*)(xr  + arow * RSX + kk * 32 + half * 8);
      uint4 a1 = *(const uint4*)(xr  + arow * RSX + kk * 32 + 16 + half * 8);
      uint4 b0 = *(const uint4*)(wvb + bcol * RSX + kk * 32 + half * 8);
      uint4 b1 = *(const uint4*)(wvb + bcol * RSX + kk * 32 + 16 + half * 8);
      A.u[0] = a0.x; A.u[1] = a0.y; A.u[2] = a0.z; A.u[3] = a0.w;
      A.u[4] = a1.x; A.u[5] = a1.y; A.u[6] = a1.z; A.u[7] = a1.w;
      Bm.u[0] = b0.x; Bm.u[1] = b0.y; Bm.u[2] = b0.z; Bm.u[3] = b0.w;
      Bm.u[4] = b1.x; Bm.u[5] = b1.y; Bm.u[6] = b1.z; Bm.u[7] = b1.w;
      acc.v = __builtin_amdgcn_wmma_f32_16x16x32_bf16(
          false, A.v, false, Bm.v, (short)0, acc.v, false, false);
    }
    uint4 p;
    p.x = pkbf(acc.f[0], acc.f[1]); p.y = pkbf(acc.f[2], acc.f[3]);
    p.z = pkbf(acc.f[4], acc.f[5]); p.w = pkbf(acc.f[6], acc.f[7]);
    *(uint4*)(vt + bcol * RSVT + nb + half * 8) = p;
  }
  // ---- Phase 1b: s_src / s_dst (rank-1 score components), vectorized xr reads
  for (int idx = tid; idx < 800; idx += 256) {
    int n = idx >> 1, h = idx & 1;
    float ss = 0.f, sd = 0.f;
    #pragma unroll
    for (int cw = 0; cw < 8; ++cw) {
      uint4 xv4 = *(const uint4*)(xr + n * RSX + cw * 8);
      uint32_t uu[4] = {xv4.x, xv4.y, xv4.z, xv4.w};
      #pragma unroll
      for (int z = 0; z < 4; ++z) {
        float x0 = bf2f(uu[z] & 0xFFFFu), x1 = bf2f(uu[z] >> 16);
        int c = cw * 8 + z * 2;
        ss += x0 * weff[h * 64 + c]     + x1 * weff[h * 64 + c + 1];
        sd += x0 * weff[128 + h * 64 + c] + x1 * weff[128 + h * 64 + c + 1];
      }
    }
    ssrc[h * 400 + n] = ss;
    sdstp[h * 416 + n] = sd;
  }
  __syncthreads();

  // ---- Phase 2: pack gso into LDS bitmask (overwrites dead xr region) + maH
  if (tid < 2) {
    float ma = -3.0e38f;
    for (int j = 0; j < 400; ++j) ma = fmaxf(ma, sdstp[tid * 416 + j]);
    maH[tid] = ma;
  }
  for (int w = tid; w < 400 * 16; w += 256) {
    int i = w >> 4, wj = w & 15;
    uint32_t bits = 0;
    int jb = wj * 32;
    if (jb < 400) {
      int ng = (400 - jb) >> 2;  if (ng > 8) ng = 8;   // int4 groups (tail word: 4)
      const int4* gp = (const int4*)(gso + i * 400 + jb);
      for (int q = 0; q < ng; ++q) {
        int4 gv = gp[q];
        bits |= (uint32_t)(gv.x != 0) << (q * 4 + 0);
        bits |= (uint32_t)(gv.y != 0) << (q * 4 + 1);
        bits |= (uint32_t)(gv.z != 0) << (q * 4 + 2);
        bits |= (uint32_t)(gv.w != 0) << (q * 4 + 3);
      }
    }
    maskw[w] = bits;
  }
  __syncthreads();

  // ---- Phase 3: row max only (no exp!).  lrelu monotonic =>
  //      m = any-unmasked ? lrelu(ss + max_{unmasked j} sd_j) : lrelu(ss + max_j sd_j) - 1e9
  for (int idx = tid; idx < 800; idx += 256) {
    int i = idx >> 1, h = idx & 1;
    const float4* sp = (const float4*)(sdstp + h * 416);
    float mu = -3.0e38f;
    for (int kt = 0; kt < 13; ++kt) {
      uint32_t mw = maskw[i * 16 + kt];
      #pragma unroll
      for (int q = 0; q < 8; ++q) {
        float4 sv = sp[kt * 8 + q];
        if ((mw >> (q * 4 + 0)) & 1u) mu = fmaxf(mu, sv.x);
        if ((mw >> (q * 4 + 1)) & 1u) mu = fmaxf(mu, sv.y);
        if ((mw >> (q * 4 + 2)) & 1u) mu = fmaxf(mu, sv.z);
        if ((mw >> (q * 4 + 3)) & 1u) mu = fmaxf(mu, sv.w);
      }
    }
    float ss = ssrc[h * 400 + i];
    float m;
    if (mu > -1.0e37f) {
      float sc = ss + mu;  m = sc > 0.f ? sc : 0.2f * sc;
    } else {
      float sc = ss + maH[h];  m = (sc > 0.f ? sc : 0.2f * sc) - 1e9f;
    }
    rowm[h * 400 + i] = m;
  }
  __syncthreads();

  // ---- Phase 4: out = attn @ V via WMMA; softmax denom accumulated from A tiles;
  //      fused LayerNorm; coalesced float4 writeout
  float gam4[4], bet4[4];
  #pragma unroll
  for (int k = 0; k < 4; ++k) { gam4[k] = gamma[k * 16 + l15]; bet4[k] = beta[k * 16 + l15]; }
  float* mystage = stage + wave * (16 * 65);
  for (int rt = wave; rt < 25; rt += 8) {
    int ib = rt * 16;
    int i  = ib + l15;                 // A row held by this lane (both halves)
    F32x8 acc[4];
    #pragma unroll
    for (int k = 0; k < 4; ++k)
      #pragma unroll
      for (int r = 0; r < 8; ++r) acc[k].f[r] = 0.f;

    #pragma unroll
    for (int h = 0; h < 2; ++h) {
      float ssv = ssrc[h * 400 + i], mv = rowm[h * 400 + i];
      float asum = 0.f;                               // per-lane partial softmax denom
      const float* sp = sdstp + h * 416;
      #pragma unroll 1
      for (int kt = 0; kt < 13; ++kt) {               // K = j, 13*32 = 416 (zero-padded)
        uint32_t mw = maskw[i * 16 + kt];
        float4 c0a = *(const float4*)(sp + kt * 32 + half * 8);
        float4 c0b = *(const float4*)(sp + kt * 32 + half * 8 + 4);
        float4 c1a = *(const float4*)(sp + kt * 32 + 16 + half * 8);
        float4 c1b = *(const float4*)(sp + kt * 32 + 16 + half * 8 + 4);
        float sdv[16] = {c0a.x, c0a.y, c0a.z, c0a.w, c0b.x, c0b.y, c0b.z, c0b.w,
                         c1a.x, c1a.y, c1a.z, c1a.w, c1b.x, c1b.y, c1b.z, c1b.w};
        float av[16];
        #pragma unroll
        for (int q = 0; q < 16; ++q) {
          int off = (q < 8) ? (half * 8 + q) : (16 + half * 8 + (q - 8));
          float sc = ssv + sdv[q];
          sc = sc > 0.f ? sc : 0.2f * sc;
          if (!((mw >> off) & 1u)) sc -= 1e9f;        // masked -> exp underflows to 0
          float e = __expf(sc - mv);                  // unnormalized
          av[q] = e;  asum += e;
        }
        BF16x16 A;
        #pragma unroll
        for (int w2 = 0; w2 < 8; ++w2) A.u[w2] = pkbf(av[2 * w2], av[2 * w2 + 1]);
        #pragma unroll
        for (int et = 0; et < 2; ++et) {
          int ch = h * 32 + et * 16 + l15;            // B column = output channel
          BF16x16 Bm;
          uint4 b0 = *(const uint4*)(vt + ch * RSVT + kt * 32 + half * 8);
          uint4 b1 = *(const uint4*)(vt + ch * RSVT + kt * 32 + 16 + half * 8);
          Bm.u[0] = b0.x; Bm.u[1] = b0.y; Bm.u[2] = b0.z; Bm.u[3] = b0.w;
          Bm.u[4] = b1.x; Bm.u[5] = b1.y; Bm.u[6] = b1.z; Bm.u[7] = b1.w;
          acc[h * 2 + et].v = __builtin_amdgcn_wmma_f32_16x16x32_bf16(
              false, A.v, false, Bm.v, (short)0, acc[h * 2 + et].v, false, false);
        }
      }
      // softmax denominator for row i (this head): combine the two j-halves
      float rl = 1.0f / (asum + __shfl_xor(asum, 16, 32));
      // redistribute from A-layout (row=l15) to D-layout (row = r + 8*half)
      #pragma unroll
      for (int r = 0; r < 8; ++r) {
        float rlr = __shfl(rl, r + 8 * half, 32);
        acc[h * 2 + 0].f[r] *= rlr;
        acc[h * 2 + 1].f[r] *= rlr;
      }
    }
    // LayerNorm over 64 channels: acc[k].f[r] = out(i=ib+r+8*half, ch=k*16+l15)
    #pragma unroll
    for (int r = 0; r < 8; ++r) {
      float s1 = acc[0].f[r] + acc[1].f[r] + acc[2].f[r] + acc[3].f[r];
      float s2 = acc[0].f[r] * acc[0].f[r] + acc[1].f[r] * acc[1].f[r]
               + acc[2].f[r] * acc[2].f[r] + acc[3].f[r] * acc[3].f[r];
      #pragma unroll
      for (int msk = 1; msk < 16; msk <<= 1) {
        s1 += __shfl_xor(s1, msk, 32);
        s2 += __shfl_xor(s2, msk, 32);
      }
      float mean = s1 * (1.f / 64.f);
      float var  = s2 * (1.f / 64.f) - mean * mean;
      float rs   = rsqrtf(var + 1e-5f);
      int il = r + 8 * half;
      #pragma unroll
      for (int k = 0; k < 4; ++k) {
        int ch = k * 16 + l15;
        mystage[il * 65 + ch] = (acc[k].f[r] - mean) * rs * gam4[k] + bet4[k];
      }
    }
    asm volatile("s_wait_dscnt 0" ::: "memory");     // wave-private stage handoff
    // writeout: float4 per lane, 64B per 4 lanes per channel row
    for (int u = lane; u < 256; u += 32) {
      int ch = u >> 2, q = u & 3;
      float4 o;
      o.x = mystage[(q * 4 + 0) * 65 + ch];
      o.y = mystage[(q * 4 + 1) * 65 + ch];
      o.z = mystage[(q * 4 + 2) * 65 + ch];
      o.w = mystage[(q * 4 + 3) * 65 + ch];
      *(float4*)(out + ((size_t)(b * 64 + ch) * 32 + t) * 400 + ib + q * 4) = o;
    }
  }
}

extern "C" void kernel_launch(void* const* d_in, const int* in_sizes, int n_in,
                              void* d_out, int out_size, void* d_ws, size_t ws_size,
                              hipStream_t stream) {
  const float* x     = (const float*)d_in[0];
  const float* Wq    = (const float*)d_in[1];
  const float* Wk    = (const float*)d_in[2];
  const float* Wv    = (const float*)d_in[3];
  const float* a_src = (const float*)d_in[4];
  const float* a_dst = (const float*)d_in[5];
  const float* gamma = (const float*)d_in[6];
  const float* beta  = (const float*)d_in[7];
  const int*   gso   = (const int*)d_in[8];
  float* out = (float*)d_out;
  (void)in_sizes; (void)n_in; (void)out_size; (void)d_ws; (void)ws_size;

  hipFuncSetAttribute((const void*)gat_fused,
                      hipFuncAttributeMaxDynamicSharedMemorySize, SMEM_BYTES);
  gat_fused<<<dim3(256), dim3(256), SMEM_BYTES, stream>>>(
      x, Wq, Wk, Wv, a_src, a_dst, gamma, beta, gso, out);
}
